// MaskedQRNNLayer_24790551232655
// MI455X (gfx1250) — compile-verified, matrix operand order checked by reading, not run on hardware
//
#include <hip/hip_runtime.h>
#include <hip/hip_bf16.h>
#include <stdint.h>

// MaskedQRNN fused kernel for gfx1250 (MI455X).
// GEMM in split-precision bf16 (hi+lo planes, 3 independent WMMA accumulators,
// f32 acc), fused with activations and the sequential scan. Zero workspace.
// 6 waves/block: gate g in {Z,F,O} x K-half; partials reduced in LDS.
// X slab for step s+1 is prefetched (global_prefetch) while step s computes.

#define SEQ  512
#define NB   64
#define DIN  1024
#define DH   1024
#define WS   (DIN + 8)     // padded LDS row stride (elements) to break bank conflicts
#define NT   192           // 6 waves
#define KHALF (DIN / 2)

typedef __attribute__((ext_vector_type(16))) __bf16 v16bf;
typedef __attribute__((ext_vector_type(8)))  float  v8f;

union FragB16 {
    v16bf v;
    uint4 q[2];
};

union Half4 {
    __bf16 b[4];
    uint2  u;
};

__device__ __forceinline__ float sigmoidf_fast(float x) {
    return 1.0f / (1.0f + __expf(-x));
}
__device__ __forceinline__ float tanhf_fast(float x) {
    // tanh(x) = 1 - 2/(exp(2x)+1); saturates correctly at +/-inf
    return 1.0f - 2.0f / (__expf(2.0f * x) + 1.0f);
}

__device__ __forceinline__ void split4(const float4 x, Half4& h, Half4& l) {
    h.b[0] = (__bf16)x.x;  h.b[1] = (__bf16)x.y;
    h.b[2] = (__bf16)x.z;  h.b[3] = (__bf16)x.w;
    l.b[0] = (__bf16)(x.x - (float)h.b[0]);
    l.b[1] = (__bf16)(x.y - (float)h.b[1]);
    l.b[2] = (__bf16)(x.z - (float)h.b[2]);
    l.b[3] = (__bf16)(x.w - (float)h.b[3]);
}

extern "C" __global__ __launch_bounds__(NT)
void qrnn_fused_kernel(const float* __restrict__ X,
                       const float* __restrict__ seqmask,
                       const float* __restrict__ hidden,
                       const float* __restrict__ W,
                       const float* __restrict__ bias,
                       float* __restrict__ out)
{
    extern __shared__ char smem_raw[];
    __bf16* w_hi = (__bf16*)smem_raw;                 // [48][WS]
    __bf16* w_lo = w_hi + 48 * WS;                    // [48][WS]
    __bf16* x_hi = w_lo + 48 * WS;                    // [16][WS]
    __bf16* x_lo = x_hi + 16 * WS;                    // [16][WS]
    float*  stage   = (float*)(x_lo + 16 * WS);       // [6][256] gate partials
    float*  c_state = stage + 6 * 256;                // [256] carry
    float*  b_s     = c_state + 256;                  // [48] bias tile

    const int tid    = threadIdx.x;
    const int wavew  = tid >> 5;
    const int gate   = wavew >> 1;      // 0=Z, 1=F, 2=O
    const int khalf  = wavew & 1;       // K range [khalf*512, khalf*512+512)
    const int lane   = tid & 31;
    const int nIdx   = lane & 15;       // A-row m / B-col n / C-col n
    const int hiHalf = lane >> 4;
    const int kbase  = hiHalf * 8;      // 16-bit A/B layout: lanes 16-31 shift K by 8

    const int h0 = blockIdx.x * 16;     // hidden tile
    const int b0 = blockIdx.y * 16;     // batch tile

    // ---- preload this block's 48 W rows, split into bf16 hi/lo planes ----
    for (int i = tid; i < 48 * (DIN / 4); i += NT) {
        int row = i >> 8;               // 0..47  (gate*16 + local row)
        int d   = (i & 255) * 4;
        int g   = row >> 4;
        int r   = row & 15;
        float4 w = *(const float4*)(W + ((size_t)(g * DH + h0 + r)) * DIN + d);
        Half4 h, l;
        split4(w, h, l);
        *(uint2*)(w_hi + row * WS + d) = h.u;
        *(uint2*)(w_lo + row * WS + d) = l.u;
    }
    // ---- init carry from hidden; bias tile ----
    for (int i = tid; i < 256; i += NT) {
        int m = i >> 4, n = i & 15;
        c_state[i] = hidden[(size_t)(b0 + m) * DH + h0 + n];
    }
    if (tid < 48) {
        b_s[tid] = bias[(tid >> 4) * DH + h0 + (tid & 15)];
    }
    __syncthreads();

    // Per-plane fragment base pointers: runtime part folded in once, so every
    // LDS access in the (fully unrolled) K loop is base + 16-bit imm offset.
    const int kofs = khalf * KHALF + kbase;
    const __bf16* pah = x_hi + nIdx * WS + kofs;
    const __bf16* pal = x_lo + nIdx * WS + kofs;
    const __bf16* pbh = w_hi + (gate * 16 + nIdx) * WS + kofs;
    const __bf16* pbl = w_lo + (gate * 16 + nIdx) * WS + kofs;
    float* my_stage = stage + wavew * 256;

    for (int s = 0; s < SEQ; ++s) {
        // ---- stage X[s, b0:b0+16, :] into bf16 hi/lo planes (4-wide) ----
        const float* xsrc = X + ((size_t)s * NB + b0) * DIN;
        for (int i = tid; i < 16 * (DIN / 4); i += NT) {
            int m = i >> 8;             // 256 float4 per row
            int d = (i & 255) * 4;
            float4 x = *(const float4*)(xsrc + (size_t)m * DIN + d);
            Half4 h, l;
            split4(x, h, l);
            *(uint2*)(x_hi + m * WS + d) = h.u;
            *(uint2*)(x_lo + m * WS + d) = l.u;
        }
        // ---- prefetch next step's X slab (overlaps with this step's WMMA) ----
        if (s + 1 < SEQ) {
            const char* nx = (const char*)(X + ((size_t)(s + 1) * NB + b0) * DIN);
            #pragma unroll
            for (int off = 0; off < 16 * DIN * 4; off += NT * 128) {
                __builtin_prefetch(nx + off + tid * 128, 0, 1);
            }
        }
        __syncthreads();

        // Three independent accumulators -> no WMMA->WMMA RAW chain per k-step.
        v8f acc_hh = {}, acc_hl = {}, acc_lh = {};
        #pragma unroll
        for (int kk = 0; kk < KHALF; kk += 32) {
            FragB16 ah, al, bh, bl;
            ah.q[0] = *(const uint4*)(pah + kk);
            ah.q[1] = *(const uint4*)(pah + kk + 16);
            al.q[0] = *(const uint4*)(pal + kk);
            al.q[1] = *(const uint4*)(pal + kk + 16);
            bh.q[0] = *(const uint4*)(pbh + kk);
            bh.q[1] = *(const uint4*)(pbh + kk + 16);
            bl.q[0] = *(const uint4*)(pbl + kk);
            bl.q[1] = *(const uint4*)(pbl + kk + 16);

            acc_hh = __builtin_amdgcn_wmma_f32_16x16x32_bf16(false, ah.v, false, bh.v,
                                                             (short)0, acc_hh, false, false);
            acc_hl = __builtin_amdgcn_wmma_f32_16x16x32_bf16(false, ah.v, false, bl.v,
                                                             (short)0, acc_hl, false, false);
            acc_lh = __builtin_amdgcn_wmma_f32_16x16x32_bf16(false, al.v, false, bh.v,
                                                             (short)0, acc_lh, false, false);
        }

        // ---- write this wave's raw 16x16 partial to staging ----
        #pragma unroll
        for (int r = 0; r < 8; ++r) {
            int m = r + hiHalf * 8;          // C/D layout: VGPR r -> row m
            my_stage[m * 16 + nIdx] = acc_hh[r] + acc_hl[r] + acc_lh[r];
        }
        __syncthreads();

        // ---- reduce K-halves, bias + activations, recurrence, output ----
        for (int i = tid; i < 256; i += NT) {
            int m = i >> 4, n = i & 15;
            float msk = seqmask[s * NB + b0 + m];
            float zy = stage[0 * 256 + i] + stage[1 * 256 + i] + b_s[n];
            float fy = stage[2 * 256 + i] + stage[3 * 256 + i] + b_s[16 + n];
            float oy = stage[4 * 256 + i] + stage[5 * 256 + i] + b_s[32 + n];
            float z = tanhf_fast(zy);
            float f = sigmoidf_fast(fy) * msk;
            float c = f * z + (1.0f - f) * c_state[i];
            c_state[i] = c;
            out[((size_t)s * NB + b0 + m) * DH + h0 + n] = sigmoidf_fast(oy) * c * msk;
        }
        __syncthreads();
    }

    // ---- final carry C[-1] ----
    float* outC = out + (size_t)SEQ * NB * DH;
    for (int i = tid; i < 256; i += NT) {
        int m = i >> 4, n = i & 15;
        outC[(size_t)(b0 + m) * DH + h0 + n] = c_state[i];
    }
}

extern "C" void kernel_launch(void* const* d_in, const int* in_sizes, int n_in,
                              void* d_out, int out_size, void* d_ws, size_t ws_size,
                              hipStream_t stream) {
    (void)in_sizes; (void)n_in; (void)out_size; (void)d_ws; (void)ws_size;
    const float* X       = (const float*)d_in[0];
    const float* seqmask = (const float*)d_in[1];
    const float* hidden  = (const float*)d_in[2];
    const float* W       = (const float*)d_in[3];
    const float* b       = (const float*)d_in[4];
    float* out = (float*)d_out;

    // LDS: W hi/lo planes + X hi/lo planes + partial stage + carry + bias
    size_t lds_bytes = (size_t)(48 * WS) * 2 * sizeof(__bf16)   // W planes
                     + (size_t)(16 * WS) * 2 * sizeof(__bf16)   // X planes
                     + (size_t)(6 * 256) * sizeof(float)        // partials
                     + (size_t)256 * sizeof(float)              // carry
                     + (size_t)48 * sizeof(float);              // bias

    dim3 grid(DH / 16, NB / 16, 1);   // 64 hidden tiles x 4 batch tiles = 256 blocks
    dim3 block(NT, 1, 1);             // 6 waves: {Z,F,O} x {K-half 0,1}
    qrnn_fused_kernel<<<grid, block, lds_bytes, stream>>>(X, seqmask, hidden, W, b, out);
}
